// PerNodeLSTMEncoder_40767829573726
// MI455X (gfx1250) — matrix operand order repaired and checked by reading, not compiled, not run on hardware
//
#include <hip/hip_runtime.h>
#include <hip/hip_bf16.h>
#include <cstdint>
#include <cstddef>

// ---------------------------------------------------------------------------
// Problem constants (from reference): N=128, T=512, I=32, H=256, G=4H=1024
// ---------------------------------------------------------------------------
static constexpr int N_ = 128;
static constexpr int T_ = 512;
static constexpr int I_ = 32;
static constexpr int H_ = 256;
static constexpr int G_ = 1024;          // 4*H
static constexpr int MT_ = T_ / 16;      // 32  M-tiles (time)
static constexpr int GT_ = G_ / 16;      // 64  N-tiles (gates)
static constexpr int NK0_ = I_ / 32;     // 1   K-steps layer0
static constexpr int NK1_ = H_ / 32;     // 8   K-steps layer1

typedef _Float16 v16h __attribute__((ext_vector_type(16)));
typedef _Float16 v2h  __attribute__((ext_vector_type(2)));
typedef float    v8f  __attribute__((ext_vector_type(8)));

// ---------------------------------------------------------------------------
// Workspace layout (bytes). Total ~493 MB.
// ---------------------------------------------------------------------------
static constexpr size_t OFF_XW     = 0;                                   // f32 [N][T][G]   256 MB (reused by both layers)
static constexpr size_t OFF_BIAS0  = OFF_XW     + (size_t)N_*T_*G_*4;     // f32 [N][G]
static constexpr size_t OFF_BIAS1  = OFF_BIAS0  + (size_t)N_*G_*4;
static constexpr size_t OFF_HLAST  = OFF_BIAS1  + (size_t)N_*G_*4;        // f32 [N][H]
static constexpr size_t OFF_AFRAG0 = OFF_HLAST  + (size_t)N_*H_*4;        // f16 swizzled x     [N][MT][1][32][16]
static constexpr size_t OFF_BFRAG0 = OFF_AFRAG0 + (size_t)N_*T_*I_*2;     // f16 swizzled Wih0  [N][GT][1][32][16]
static constexpr size_t OFF_BFRAG1 = OFF_BFRAG0 + (size_t)N_*G_*I_*2;     // f16 swizzled Wih1  [N][GT][8][32][16]
static constexpr size_t OFF_AFRAG1 = OFF_BFRAG1 + (size_t)N_*G_*H_*2;     // f16 swizzled hs0   [N][MT][8][32][16]
static constexpr size_t OFF_WHHT0  = OFF_AFRAG1 + (size_t)N_*T_*H_*2;     // f16-pairs [N][H/2][G][2]
static constexpr size_t OFF_WHHT1  = OFF_WHHT0  + (size_t)N_*H_*G_*2;

// ---------------------------------------------------------------------------
// WMMA per-lane fragment mappings (ISA 7.12.2, 16-bit A 16x32 / B 32x16)
// A: lane<16  holds row m=lane,    K = slot<8 ? slot    : slot+8
//    lane>=16 holds row m=lane-16, K = slot<8 ? slot+8  : slot+16
// B: lane<16  holds col j=lane,    K = slot<8 ? slot    : slot+8
//    lane>=16 holds col j=lane-16, K = slot<8 ? slot+8  : slot+16
// C/D f32 16x16: VGPR r -> row r (lanes 0-15, col=lane) / row r+8 (lanes 16-31)
// ---------------------------------------------------------------------------
__device__ __forceinline__ int a_frag_k(int lane, int slot) {
  if (lane < 16) return (slot < 8) ? slot : slot + 8;
  return (slot < 8) ? slot + 8 : slot + 16;
}

// ---------------------------------------------------------------------------
// Pack combined biases: bias = bih + bhh (both layers)
// ---------------------------------------------------------------------------
__global__ void pack_bias_kernel(const float* __restrict__ bih0, const float* __restrict__ bhh0,
                                 const float* __restrict__ bih1, const float* __restrict__ bhh1,
                                 float* __restrict__ bias0, float* __restrict__ bias1) {
  int idx = blockIdx.x * blockDim.x + threadIdx.x;
  if (idx < N_ * G_) {
    bias0[idx] = bih0[idx] + bhh0[idx];
    bias1[idx] = bih1[idx] + bhh1[idx];
  }
}

// ---------------------------------------------------------------------------
// Pack x (f32 [N][T][I]) into A-fragments (f16), layer-0 (single K-step).
// ---------------------------------------------------------------------------
__global__ void pack_a_x_kernel(const float* __restrict__ x, _Float16* __restrict__ afrag) {
  long long o = (long long)blockIdx.x * blockDim.x + threadIdx.x;   // one half per thread
  if (o >= (long long)N_ * T_ * I_) return;
  int slot = (int)(o & 15);
  int lane = (int)((o >> 4) & 31);
  int mt   = (int)((o >> 9) & (MT_ - 1));
  int n    = (int)(o >> 14);
  int m = lane & 15;
  int k = a_frag_k(lane, slot);
  afrag[o] = (_Float16)x[((size_t)n * T_ + mt * 16 + m) * I_ + k];
}

// ---------------------------------------------------------------------------
// Pack Wih (f32 [N][G][K]) into B-fragments (f16).  B element (k, j) = Wih[g0+j][k].
// ---------------------------------------------------------------------------
__global__ void pack_b_kernel(const float* __restrict__ w, _Float16* __restrict__ bfrag, int nk, int kdim) {
  long long o = (long long)blockIdx.x * blockDim.x + threadIdx.x;
  long long total = (long long)N_ * GT_ * nk * 512;
  if (o >= total) return;
  int slot = (int)(o & 15);
  int lane = (int)((o >> 4) & 31);
  long long tile = o >> 9;
  int kt = (int)(tile % nk);
  int gt = (int)((tile / nk) % GT_);
  int n  = (int)(tile / ((long long)nk * GT_));
  int j  = lane & 15;
  int kg = kt * 32 + ((slot < 8) ? slot : slot + 8) + ((lane >= 16) ? 8 : 0);
  bfrag[o] = (_Float16)w[((size_t)n * G_ + gt * 16 + j) * kdim + kg];
}

// ---------------------------------------------------------------------------
// Transpose Whh (f32 [N][G][H]) -> f16-pair layout [N][H/2][G] (uint = 2 halfs)
// so the recurrence reads one coalesced uint per (k-pair, gate-output).
// ---------------------------------------------------------------------------
__global__ void pack_whht_kernel(const float* __restrict__ whh, uint32_t* __restrict__ out) {
  long long o = (long long)blockIdx.x * blockDim.x + threadIdx.x;
  if (o >= (long long)N_ * (H_ / 2) * G_) return;
  int g  = (int)(o & (G_ - 1));
  int hp = (int)((o >> 10) & (H_ / 2 - 1));
  int n  = (int)(o >> 17);
  const float* src = whh + ((size_t)n * G_ + g) * H_ + 2 * hp;
  v2h p; p[0] = (_Float16)src[0]; p[1] = (_Float16)src[1];
  out[o] = __builtin_bit_cast(uint32_t, p);
}

// ---------------------------------------------------------------------------
// Batched GEMM via WMMA: xW[n][t][g] = sum_k A[n][t][k] * Wih[n][g][k] + bias[n][g]
// grid = (MT, GT/8, N), block = 256 (8 waves, one 16x16 tile per wave)
// ---------------------------------------------------------------------------
__global__ __launch_bounds__(256) void gemm_xw_kernel(const _Float16* __restrict__ afrag,
                                                      const _Float16* __restrict__ bfrag,
                                                      const float* __restrict__ bias,
                                                      float* __restrict__ xw, int nk) {
  const int lane = threadIdx.x & 31;
  const int wave = threadIdx.x >> 5;
  const int mt = blockIdx.x;
  const int gt = blockIdx.y * 8 + wave;
  const int n  = blockIdx.z;

  v8f c = {};
  for (int kt = 0; kt < nk; ++kt) {
    const v16h a = *(const v16h*)(afrag + ((((size_t)n * MT_ + mt) * nk + kt) << 9) + (size_t)lane * 16);
    const v16h b = *(const v16h*)(bfrag + ((((size_t)n * GT_ + gt) * nk + kt) << 9) + (size_t)lane * 16);
    c = __builtin_amdgcn_wmma_f32_16x16x32_f16(false, a, false, b, (short)0, c, false, false);
  }

  const int col = gt * 16 + (lane & 15);
  const float bv = bias[(size_t)n * G_ + col];
  const int rbase = (lane >= 16) ? 8 : 0;
  float* outp = xw + ((size_t)n * T_ + mt * 16) * G_ + col;
#pragma unroll
  for (int r = 0; r < 8; ++r) {
    outp[(size_t)(r + rbase) * G_] = c[r] + bv;
  }
}

// ---------------------------------------------------------------------------
// Sequential LSTM recurrence. One workgroup per node, 1024 threads
// (thread j owns gate-output j; gate = j>>8 in PyTorch order i,f,g,o).
// Whh streamed from L2 as coalesced f16 pairs; h broadcast from LDS;
// v_dot2_f32_f16 when available.
// layer==0: writes h_t into layer-1 A-fragment layout (f16).
// layer==1: writes final h into hlast (f32).
// ---------------------------------------------------------------------------
__global__ __launch_bounds__(1024) void lstm_rec_kernel(const float* __restrict__ xw,
                                                        const uint32_t* __restrict__ whht,
                                                        _Float16* __restrict__ afrag1,
                                                        float* __restrict__ hlast,
                                                        int layer) {
  const int n = blockIdx.x;
  const int j = threadIdx.x;            // 0..1023

  __shared__ uint32_t hbuf[H_ / 2];     // h as packed f16 pairs
  __shared__ float    act[G_];          // activated gates

  if (j < H_ / 2) hbuf[j] = 0u;
  __syncthreads();

  float c = 0.0f;                       // cell state (threads j<H only)
  const uint32_t* wcol = whht + (size_t)n * (H_ / 2) * G_ + j;
  const float* xrow = xw + (size_t)n * T_ * G_ + j;

  for (int t = 0; t < T_; ++t) {
    float acc = xrow[(size_t)t * G_];
    if (t + 1 < T_) __builtin_prefetch(&xrow[(size_t)(t + 1) * G_], 0, 1);  // global_prefetch_b8

#pragma unroll 8
    for (int hp = 0; hp < H_ / 2; ++hp) {
      uint32_t w  = wcol[(size_t)hp * G_];
      uint32_t hv = hbuf[hp];
#if __has_builtin(__builtin_amdgcn_fdot2)
      acc = __builtin_amdgcn_fdot2(__builtin_bit_cast(v2h, w),
                                   __builtin_bit_cast(v2h, hv), acc, false);
#else
      v2h wv = __builtin_bit_cast(v2h, w);
      v2h hh = __builtin_bit_cast(v2h, hv);
      acc += (float)wv[0] * (float)hh[0];
      acc += (float)wv[1] * (float)hh[1];
#endif
    }

    const int gate = j >> 8;
    float v = (gate == 2) ? tanhf(acc) : (1.0f / (1.0f + __expf(-acc)));
    act[j] = v;
    __syncthreads();

    if (j < H_) {
      float ig = act[j];
      float fg = act[j + H_];
      float gg = act[j + 2 * H_];
      float og = act[j + 3 * H_];
      c = fg * c + ig * gg;
      float h = og * tanhf(c);
      reinterpret_cast<_Float16*>(hbuf)[j] = (_Float16)h;
      if (layer == 0) {
        // scatter into layer-1 A-fragment layout: (n, mt=t/16, kt=j/32, lane, slot)
        int kk = j & 31, m = t & 15, mt = t >> 4, kt = j >> 5;
        int lane, slot;
        if (kk < 8)       { lane = m;      slot = kk;      }
        else if (kk < 16) { lane = m + 16; slot = kk - 8;  }
        else if (kk < 24) { lane = m;      slot = kk - 8;  }
        else              { lane = m + 16; slot = kk - 16; }
        afrag1[((((size_t)n * MT_ + mt) * NK1_ + kt) * 32 + lane) * 16 + slot] = (_Float16)h;
      } else if (t == T_ - 1) {
        hlast[(size_t)n * H_ + j] = h;
      }
    }
    __syncthreads();   // h_t fully visible before next step's dot products
  }
}

// ---------------------------------------------------------------------------
// LayerNorm over H=256 of final hidden state. One block per node row.
// ---------------------------------------------------------------------------
__global__ __launch_bounds__(256) void layernorm_kernel(const float* __restrict__ hlast,
                                                        const float* __restrict__ gamma,
                                                        const float* __restrict__ beta,
                                                        float* __restrict__ out) {
  const int n = blockIdx.x;
  const int j = threadIdx.x;
  __shared__ float red[256];

  float v = hlast[(size_t)n * H_ + j];
  red[j] = v;
  __syncthreads();
  for (int s = 128; s > 0; s >>= 1) {
    if (j < s) red[j] += red[j + s];
    __syncthreads();
  }
  const float mu = red[0] * (1.0f / H_);
  __syncthreads();

  float d = v - mu;
  red[j] = d * d;
  __syncthreads();
  for (int s = 128; s > 0; s >>= 1) {
    if (j < s) red[j] += red[j + s];
    __syncthreads();
  }
  const float var = red[0] * (1.0f / H_);
  out[(size_t)n * H_ + j] = d * __frsqrt_rn(var + 1e-5f) * gamma[j] + beta[j];
}

// ---------------------------------------------------------------------------
// Launch sequence
// ---------------------------------------------------------------------------
extern "C" void kernel_launch(void* const* d_in, const int* in_sizes, int n_in,
                              void* d_out, int out_size, void* d_ws, size_t ws_size,
                              hipStream_t stream) {
  const float* x     = (const float*)d_in[0];
  const float* Wih0  = (const float*)d_in[1];
  const float* Whh0  = (const float*)d_in[2];
  const float* bih0  = (const float*)d_in[3];
  const float* bhh0  = (const float*)d_in[4];
  const float* Wih1  = (const float*)d_in[5];
  const float* Whh1  = (const float*)d_in[6];
  const float* bih1  = (const float*)d_in[7];
  const float* bhh1  = (const float*)d_in[8];
  const float* gamma = (const float*)d_in[9];
  const float* beta  = (const float*)d_in[10];

  char* ws = (char*)d_ws;
  float*     xw     = (float*)(ws + OFF_XW);
  float*     bias0  = (float*)(ws + OFF_BIAS0);
  float*     bias1  = (float*)(ws + OFF_BIAS1);
  float*     hlast  = (float*)(ws + OFF_HLAST);
  _Float16*  afrag0 = (_Float16*)(ws + OFF_AFRAG0);
  _Float16*  bfrag0 = (_Float16*)(ws + OFF_BFRAG0);
  _Float16*  bfrag1 = (_Float16*)(ws + OFF_BFRAG1);
  _Float16*  afrag1 = (_Float16*)(ws + OFF_AFRAG1);
  uint32_t*  whht0  = (uint32_t*)(ws + OFF_WHHT0);
  uint32_t*  whht1  = (uint32_t*)(ws + OFF_WHHT1);

  // --- one-time packing / precision conversion ---
  pack_bias_kernel<<<(N_ * G_ + 255) / 256, 256, 0, stream>>>(bih0, bhh0, bih1, bhh1, bias0, bias1);
  pack_a_x_kernel<<<(int)(((long long)N_ * T_ * I_ + 255) / 256), 256, 0, stream>>>(x, afrag0);
  pack_b_kernel<<<(int)(((long long)N_ * GT_ * NK0_ * 512 + 255) / 256), 256, 0, stream>>>(Wih0, bfrag0, NK0_, I_);
  pack_b_kernel<<<(int)(((long long)N_ * GT_ * NK1_ * 512 + 255) / 256), 256, 0, stream>>>(Wih1, bfrag1, NK1_, H_);
  pack_whht_kernel<<<(int)(((long long)N_ * (H_ / 2) * G_ + 255) / 256), 256, 0, stream>>>(Whh0, whht0);
  pack_whht_kernel<<<(int)(((long long)N_ * (H_ / 2) * G_ + 255) / 256), 256, 0, stream>>>(Whh1, whht1);

  // --- layer 0: input projection (WMMA) + recurrence (writes layer-1 A frags) ---
  gemm_xw_kernel<<<dim3(MT_, GT_ / 8, N_), 256, 0, stream>>>(afrag0, bfrag0, bias0, xw, NK0_);
  lstm_rec_kernel<<<N_, 1024, 0, stream>>>(xw, whht0, afrag1, hlast, 0);

  // --- layer 1: input projection (WMMA, reuses xw buffer) + recurrence ---
  gemm_xw_kernel<<<dim3(MT_, GT_ / 8, N_), 256, 0, stream>>>(afrag1, bfrag1, bias1, xw, NK1_);
  lstm_rec_kernel<<<N_, 1024, 0, stream>>>(xw, whht1, afrag1, hlast, 1);

  // --- final LayerNorm on h_last ---
  layernorm_kernel<<<N_, 256, 0, stream>>>(hlast, gamma, beta, (float*)d_out);
}
